// Attention_79474074845409
// MI455X (gfx1250) — matrix operand order repaired and checked by reading, not compiled, not run on hardware
//
#include <hip/hip_runtime.h>
#include <hip/hip_bf16.h>

typedef __attribute__((ext_vector_type(16))) _Float16 v16h;
typedef __attribute__((ext_vector_type(8)))  _Float16 v8h;
typedef __attribute__((ext_vector_type(8)))  float    v8f;
typedef __attribute__((ext_vector_type(4)))  float    v4f;

#define S_LEN 2048
#define MDIM  2048
#define HDS   16
#define DDIM  128

__device__ inline v8f wmma_f16(v16h a, v16h b, v8f c) {
    return __builtin_amdgcn_wmma_f32_16x16x32_f16(false, a, false, b, (short)0, c, false, false);
}

// A-fragment (16x32 f16, MxK): lane row = lane&15, kbase = (lane>=16)?16:0.
// a[0..15] = A[row][kb + 0..15]
// Caller passes base = &A[row][chunk*32 + kb]
__device__ inline v16h load_afrag(const _Float16* base) {
    union { v16h v; v8h h[2]; } u;
    u.h[0] = *(const v8h*)(base);
    u.h[1] = *(const v8h*)(base + 8);
    return u.v;
}

// B-fragment (32x16 f16, KxN): lane col = lane&15, kbase = (lane>=16)?16:0.
// b[0..15] = B[kb + 0..15][col] -> 16 consecutive halfs when B is N-major.
__device__ inline v16h load_bfrag(const _Float16* base) {
    union { v16h v; v8h h[2]; } u;
    u.h[0] = *(const v8h*)(base);
    u.h[1] = *(const v8h*)(base + 8);
    return u.v;
}

// Fallback: gather 16 f32 and convert (distinct locals so loads can clause).
__device__ inline v16h load_bfrag_f32(const float* base) {
    v4f f0 = *(const v4f*)(base);
    v4f f1 = *(const v4f*)(base + 4);
    v4f f2 = *(const v4f*)(base + 8);
    v4f f3 = *(const v4f*)(base + 12);
    v16h r;
#pragma unroll
    for (int i = 0; i < 4; ++i) {
        r[i]      = (_Float16)f0[i];
        r[4 + i]  = (_Float16)f1[i];
        r[8 + i]  = (_Float16)f2[i];
        r[12 + i] = (_Float16)f3[i];
    }
    return r;
}

// ---------------- Kernel 0: one-shot qkvw f32 -> f16 convert ----------------
__global__ __launch_bounds__(256) void cvt_w_kernel(const float* __restrict__ wf,
                                                    _Float16* __restrict__ wh) {
    size_t i = ((size_t)blockIdx.x * 256 + threadIdx.x) * 8;
    v4f f0 = *(const v4f*)(wf + i);
    v4f f1 = *(const v4f*)(wf + i + 4);
    v8h o;
#pragma unroll
    for (int j = 0; j < 4; ++j) { o[j] = (_Float16)f0[j]; o[4 + j] = (_Float16)f1[j]; }
    *(v8h*)(wh + i) = o;
}

// ---------------- Kernel 1: LayerNorm (fp32 in, fp16 out) ----------------
__global__ __launch_bounds__(256) void ln_kernel(const float* __restrict__ x,
                                                 const float* __restrict__ g,
                                                 const float* __restrict__ b,
                                                 _Float16* __restrict__ xn) {
    int row = blockIdx.x;            // 0..4095
    int tid = threadIdx.x;           // 0..255
    const float* xr = x + (size_t)row * MDIM;
    float v[8], s = 0.f, s2 = 0.f;
#pragma unroll
    for (int i = 0; i < 8; ++i) {
        v[i] = xr[tid + i * 256];
        s += v[i]; s2 += v[i] * v[i];
    }
    __shared__ float red[256], red2[256];
    red[tid] = s; red2[tid] = s2;
    __syncthreads();
    for (int off = 128; off > 0; off >>= 1) {
        if (tid < off) { red[tid] += red[tid + off]; red2[tid] += red2[tid + off]; }
        __syncthreads();
    }
    float mu = red[0] * (1.0f / MDIM);
    float var = red2[0] * (1.0f / MDIM) - mu * mu;
    float rstd = rsqrtf(var + 1e-5f);
    _Float16* xo = xn + (size_t)row * MDIM;
#pragma unroll
    for (int i = 0; i < 8; ++i) {
        int c = tid + i * 256;
        xo[c] = (_Float16)(((v[i] - mu) * rstd) * g[c] + b[c]);
    }
}

// ------- Kernel 2: QKV GEMM (32x64 tile/wave), Q,K -> [B,H,S,D]; V -> [B,H,D,S] -------
template <bool F16W>
__global__ __launch_bounds__(128) void qkv_gemm_t(
        const _Float16* __restrict__ xn, const _Float16* __restrict__ whalf,
        const float* __restrict__ wfull, const float* __restrict__ bias,
        _Float16* __restrict__ q, _Float16* __restrict__ kk, _Float16* __restrict__ vt) {
    int wid  = blockIdx.x * 4 + (threadIdx.x >> 5);   // 0..12287
    int lane = threadIdx.x & 31;
    int col  = lane & 15, hi = lane >> 4;
    int rt = wid / 96;           // 32-token row tile, 0..127
    int ns = wid % 96;           // 64-wide n super-tile, 0..95

    const _Float16* arow0 = xn + (size_t)(rt * 32 + col) * MDIM + hi * 16;
    const _Float16* arow1 = arow0 + (size_t)16 * MDIM;

    v8f acc[2][4];
#pragma unroll
    for (int g = 0; g < 2; ++g)
#pragma unroll
        for (int t = 0; t < 4; ++t)
#pragma unroll
            for (int r = 0; r < 8; ++r) acc[g][t][r] = 0.0f;

    for (int kc = 0; kc < 64; ++kc) {
        v16h a0 = load_afrag(arow0 + kc * 32);
        v16h a1 = load_afrag(arow1 + kc * 32);
        v16h bf[4];
#pragma unroll
        for (int t = 0; t < 4; ++t) {
            size_t off = (size_t)(ns * 64 + t * 16 + col) * MDIM + kc * 32 + hi * 16;
            bf[t] = F16W ? load_bfrag(whalf + off) : load_bfrag_f32(wfull + off);
        }
#pragma unroll
        for (int t = 0; t < 4; ++t) {
            acc[0][t] = wmma_f16(a0, bf[t], acc[0][t]);
            acc[1][t] = wmma_f16(a1, bf[t], acc[1][t]);
        }
    }

    // store routing: each 16-wide N tile lies entirely in one of Q/K/V
    // (2048-boundaries are multiples of 16), so the segment test is wave-uniform.
#pragma unroll
    for (int t = 0; t < 4; ++t) {
        int nbase = ns * 64 + t * 16;    // wave-uniform
        int seg   = nbase >> 11;         // 0=Q, 1=K, 2=V (uniform)
        int nloc  = nbase & 2047;        // uniform
        int h     = nloc >> 7;           // uniform head
        int d     = (nloc & 127) + col;  // per-lane depth
        float bv  = bias[nbase + col];
        if (seg == 0) {
#pragma unroll
            for (int g = 0; g < 2; ++g)
#pragma unroll
                for (int r = 0; r < 8; ++r) {
                    int token = rt * 32 + g * 16 + r + 8 * hi;
                    int bidx = token >> 11, sidx = token & (S_LEN - 1);
                    q[(((size_t)(bidx * HDS + h)) * S_LEN + sidx) * DDIM + d] =
                        (_Float16)(acc[g][t][r] + bv);
                }
        } else if (seg == 1) {
#pragma unroll
            for (int g = 0; g < 2; ++g)
#pragma unroll
                for (int r = 0; r < 8; ++r) {
                    int token = rt * 32 + g * 16 + r + 8 * hi;
                    int bidx = token >> 11, sidx = token & (S_LEN - 1);
                    kk[(((size_t)(bidx * HDS + h)) * S_LEN + sidx) * DDIM + d] =
                        (_Float16)(acc[g][t][r] + bv);
                }
        } else {
#pragma unroll
            for (int g = 0; g < 2; ++g)
#pragma unroll
                for (int r = 0; r < 8; ++r) {
                    int token = rt * 32 + g * 16 + r + 8 * hi;
                    int bidx = token >> 11, sidx = token & (S_LEN - 1);
                    vt[(((size_t)(bidx * HDS + h)) * DDIM + d) * S_LEN + sidx] =
                        (_Float16)(acc[g][t][r] + bv);
                }
        }
    }
}

// ---------------- Kernel 3: causal flash attention, one wave per 16-query tile ----------------
__global__ __launch_bounds__(128, 1) void attn_kernel(
        const _Float16* __restrict__ q, const _Float16* __restrict__ k,
        const _Float16* __restrict__ vt, _Float16* __restrict__ ctx) {
    __shared__ _Float16 P[4][16][32];   // per-wave P tile (C-layout -> A-layout shuffle)
    int w    = threadIdx.x >> 5;
    int lane = threadIdx.x & 31;
    int col  = lane & 15, hi = lane >> 4;
    int wid  = blockIdx.x * 4 + w;      // 0..4095
    int bh   = wid >> 7;                // b*16 + h
    int qt   = wid & 127;
    int q0   = qt * 16;

    const _Float16* qb = q  + (size_t)bh * S_LEN * DDIM;
    const _Float16* kb = k  + (size_t)bh * S_LEN * DDIM;
    const _Float16* vb = vt + (size_t)bh * DDIM * S_LEN;
    const _Float16* qrow = qb + (size_t)(q0 + col) * DDIM + hi * 16;

    v8f acc[8];
#pragma unroll
    for (int t = 0; t < 8; ++t)
#pragma unroll
        for (int r = 0; r < 8; ++r) acc[t][r] = 0.0f;
    float m[8], l[8];
#pragma unroll
    for (int r = 0; r < 8; ++r) { m[r] = -1e30f; l[r] = 0.0f; }

    int nkb = (q0 + 47) >> 5;           // 32-key blocks covering keys <= q0+15
    for (int kblk = 0; kblk < nkb; ++kblk) {
        int k0 = kblk * 32;
        v8f s0, s1;
#pragma unroll
        for (int r = 0; r < 8; ++r) { s0[r] = 0.0f; s1[r] = 0.0f; }
#pragma unroll
        for (int c = 0; c < 4; ++c) {
            v16h a   = load_afrag(qrow + c * 32);       // Q tile stays hot in WGP$/L2
            v16h bk0 = load_bfrag(kb + (size_t)(k0 + col) * DDIM      + c * 32 + hi * 16);
            v16h bk1 = load_bfrag(kb + (size_t)(k0 + 16 + col) * DDIM + c * 32 + hi * 16);
            s0 = wmma_f16(a, bk0, s0);
            s1 = wmma_f16(a, bk1, s1);
        }
        // causal mask + online softmax (rows live across 16-lane groups)
        float sc[8];
#pragma unroll
        for (int r = 0; r < 8; ++r) {
            int row = q0 + r + 8 * hi;
            float e0 = ((k0 + col)      <= row) ? s0[r] : -10000.0f;
            float e1 = ((k0 + 16 + col) <= row) ? s1[r] : -10000.0f;
            float vmax = fmaxf(e0, e1);
            vmax = fmaxf(vmax, __shfl_xor(vmax, 1, 32));
            vmax = fmaxf(vmax, __shfl_xor(vmax, 2, 32));
            vmax = fmaxf(vmax, __shfl_xor(vmax, 4, 32));
            vmax = fmaxf(vmax, __shfl_xor(vmax, 8, 32));
            float mn = fmaxf(m[r], vmax);
            sc[r] = __expf(m[r] - mn);
            float p0 = __expf(e0 - mn);
            float p1 = __expf(e1 - mn);
            float rs = p0 + p1;
            rs += __shfl_xor(rs, 1, 32);
            rs += __shfl_xor(rs, 2, 32);
            rs += __shfl_xor(rs, 4, 32);
            rs += __shfl_xor(rs, 8, 32);
            l[r] = l[r] * sc[r] + rs;
            m[r] = mn;
            P[w][r + 8 * hi][col]      = (_Float16)p0;
            P[w][r + 8 * hi][col + 16] = (_Float16)p1;
        }
        // single clean rescale pass over the accumulators
#pragma unroll
        for (int t = 0; t < 8; ++t)
#pragma unroll
            for (int r = 0; r < 8; ++r) acc[t][r] *= sc[r];

        asm volatile("" ::: "memory");   // order LDS store -> LDS load (same-wave DS is in-order)
        union { v16h v; v8h h[2]; } ap;
        const _Float16* pr = &P[w][col][0];          // A row = lane&15
        ap.h[0] = *(const v8h*)(pr + hi * 16);
        ap.h[1] = *(const v8h*)(pr + hi * 16 + 8);
        asm volatile("" ::: "memory");
#pragma unroll
        for (int dc = 0; dc < 8; dc += 2) {
            v16h bv0 = load_bfrag(vb + (size_t)(dc * 16 + col) * S_LEN       + k0 + hi * 16);
            v16h bv1 = load_bfrag(vb + (size_t)((dc + 1) * 16 + col) * S_LEN + k0 + hi * 16);
            acc[dc]     = wmma_f16(ap.v, bv0, acc[dc]);
            acc[dc + 1] = wmma_f16(ap.v, bv1, acc[dc + 1]);
        }
    }

    int b = bh >> 4, h = bh & 15;
    float rl[8];
#pragma unroll
    for (int r = 0; r < 8; ++r) rl[r] = 1.0f / l[r];   // 8 reciprocals instead of 64 divides
#pragma unroll
    for (int dc = 0; dc < 8; ++dc)
#pragma unroll
        for (int r = 0; r < 8; ++r) {
            int token = q0 + r + 8 * hi;
            float o = acc[dc][r] * rl[r];
            ctx[((size_t)(b * S_LEN + token)) * MDIM + h * DDIM + dc * 16 + col] = (_Float16)o;
        }
}

// ---------------- Kernel 4: output projection (32x64 tile/wave) -> f32 ----------------
__global__ __launch_bounds__(128) void oproj_gemm(
        const _Float16* __restrict__ ctx, const _Float16* __restrict__ ow,
        const _Float16* __restrict__ ob, float* __restrict__ out) {
    int wid  = blockIdx.x * 4 + (threadIdx.x >> 5);   // 0..4095
    int lane = threadIdx.x & 31;
    int col  = lane & 15, hi = lane >> 4;
    int rt = wid >> 5;        // 32-token tile 0..127
    int ns = wid & 31;        // 64-wide n super-tile 0..31

    const _Float16* arow0 = ctx + (size_t)(rt * 32 + col) * MDIM + hi * 16;
    const _Float16* arow1 = arow0 + (size_t)16 * MDIM;

    v8f acc[2][4];
#pragma unroll
    for (int g = 0; g < 2; ++g)
#pragma unroll
        for (int t = 0; t < 4; ++t)
#pragma unroll
            for (int r = 0; r < 8; ++r) acc[g][t][r] = 0.0f;

    for (int kc = 0; kc < 64; ++kc) {
        v16h a0 = load_afrag(arow0 + kc * 32);
        v16h a1 = load_afrag(arow1 + kc * 32);
        v16h bf[4];
#pragma unroll
        for (int t = 0; t < 4; ++t) {
            size_t off = (size_t)(ns * 64 + t * 16 + col) * MDIM + kc * 32 + hi * 16;
            bf[t] = load_bfrag(ow + off);
        }
#pragma unroll
        for (int t = 0; t < 4; ++t) {
            acc[0][t] = wmma_f16(a0, bf[t], acc[0][t]);
            acc[1][t] = wmma_f16(a1, bf[t], acc[1][t]);
        }
    }

#pragma unroll
    for (int g = 0; g < 2; ++g)
#pragma unroll
        for (int t = 0; t < 4; ++t) {
            int n = ns * 64 + t * 16 + col;
            float bv = (float)ob[n];
#pragma unroll
            for (int r = 0; r < 8; ++r) {
                int token = rt * 32 + g * 16 + r + 8 * hi;
                out[(size_t)token * MDIM + n] = acc[g][t][r] + bv;
            }
        }
}

extern "C" void kernel_launch(void* const* d_in, const int* in_sizes, int n_in,
                              void* d_out, int out_size, void* d_ws, size_t ws_size,
                              hipStream_t stream) {
    const float*    x    = (const float*)d_in[0];
    const float*    ln_g = (const float*)d_in[1];
    const float*    ln_b = (const float*)d_in[2];
    const float*    qkvw = (const float*)d_in[3];
    const float*    qkvb = (const float*)d_in[4];
    const _Float16* ow   = (const _Float16*)d_in[5];
    const _Float16* ob   = (const _Float16*)d_in[6];
    float* out = (float*)d_out;

    const size_t NTOK = 2 * S_LEN;                 // 4096
    const size_t TILE = NTOK * MDIM;               // 8,388,608 halfs
    _Float16* xn = (_Float16*)d_ws;                // reused as ctx after qkv_gemm
    _Float16* q  = xn + TILE;
    _Float16* kk = q  + TILE;
    _Float16* vt = kk + TILE;
    _Float16* wh = vt + TILE;                      // optional f16 copy of qkvw
    _Float16* ctx = xn;

    const size_t base_bytes = 4 * TILE * sizeof(_Float16);                // 64 MiB
    const size_t wh_bytes   = (size_t)3 * MDIM * MDIM * sizeof(_Float16); // 24 MiB
    bool preconv = ws_size >= base_bytes + wh_bytes;

    ln_kernel<<<4096, 256, 0, stream>>>(x, ln_g, ln_b, xn);
    if (preconv) {
        cvt_w_kernel<<<6144, 256, 0, stream>>>(qkvw, wh);             // 12.58M elems, 8/thread
        qkv_gemm_t<true><<<3072, 128, 0, stream>>>(xn, wh, qkvw, qkvb, q, kk, vt);
    } else {
        qkv_gemm_t<false><<<3072, 128, 0, stream>>>(xn, wh, qkvw, qkvb, q, kk, vt);
    }
    attn_kernel<<<1024, 128, 0, stream>>>(q, kk, vt, ctx);            // 4096 waves
    oproj_gemm<<<1024, 128, 0, stream>>>(ctx, ow, ob, out);           // 128*32 waves
}
